// Merge_24300924961395
// MI455X (gfx1250) — compile-verified, matrix operand order checked by reading, not run on hardware
//
#include <hip/hip_runtime.h>
#include <hip/hip_bf16.h>

// ---------------- problem constants (from reference) ----------------
#define BATCHES 8
#define T_TOK   8192
#define CH      128
#define HALF_T  4096          // T/2 (even / odd token counts)
#define R_MERGE 2048          // r = min(T - NPOINT, T/2)
#define UNM_CNT 2048          // HALF_T - R_MERGE
#define OUT_T   6144          // NPOINT
#define NTILES  (HALF_T / 16) // 256 N-tiles of 16 columns

typedef __attribute__((ext_vector_type(16))) _Float16 v16h;
typedef __attribute__((ext_vector_type(8)))  _Float16 v8h;
typedef __attribute__((ext_vector_type(8)))  float    v8f;

// Load one lane's share of a 16x32 (rows x K) fp16 operand tile in the CDNA5
// WMMA register layout:
//   lanes 0-15  (h=0): VGPR0-3 = K[base+0..7],  VGPR4-7 = K[base+16..23]
//   lanes 16-31 (h=1): VGPR0-3 = K[base+8..15], VGPR4-7 = K[base+24..31]
// row_ptr points at K=0 of the row this lane owns (lane & 15).
__device__ __forceinline__ v16h load_opnd(const _Float16* __restrict__ row_ptr,
                                          int kc, int h) {
  const _Float16* p = row_ptr + kc * 32 + h * 8;
  v8h lo = *(const v8h*)(p);        // 16B
  v8h hi = *(const v8h*)(p + 16);   // 16B
  return __builtin_shufflevector(lo, hi,
                                 0, 1, 2, 3, 4, 5, 6, 7,
                                 8, 9, 10, 11, 12, 13, 14, 15);
}

// ---------------- 1) metric = x / ||x||, split even/odd, fp16 ----------------
__global__ void __launch_bounds__(256)
normalize_kernel(const float* __restrict__ pts,
                 _Float16* __restrict__ am, _Float16* __restrict__ bm) {
  const int lane = threadIdx.x & 31;
  const int wave = threadIdx.x >> 5;
  const size_t tok = (size_t)blockIdx.x * 8 + wave;   // 0 .. B*T-1
  const int b = (int)(tok >> 13);                     // /8192
  const int t = (int)(tok & 8191);

  const float4 v = ((const float4*)(pts + tok * CH))[lane];
  float ss = v.x * v.x + v.y * v.y + v.z * v.z + v.w * v.w;
  #pragma unroll
  for (int off = 1; off < 32; off <<= 1) ss += __shfl_xor(ss, off, 32);
  const float inv = rsqrtf(ss);

  _Float16* dst = ((t & 1) ? bm : am) +
                  ((size_t)b * HALF_T + (t >> 1)) * CH + lane * 4;
  dst[0] = (_Float16)(v.x * inv);
  dst[1] = (_Float16)(v.y * inv);
  dst[2] = (_Float16)(v.z * inv);
  dst[3] = (_Float16)(v.w * inv);
}

// ---------------- 2) fused scores = A·Bᵀ  +  row max/argmax ----------------
// One wave owns a 16-row A strip; loops over 256 B tiles; 4 WMMAs per tile
// (K = 128 = 4 x 32). B tiles are register double-buffered so tile nt+1's
// eight b128 loads issue while tile nt's WMMAs execute. Row max/argmax kept
// in registers, reduced by a 16-lane butterfly at the end (ties -> lower
// column index, matching jnp.argmax first-occurrence).
__global__ void __launch_bounds__(256)
score_argmax_kernel(const _Float16* __restrict__ am,
                    const _Float16* __restrict__ bm,
                    float* __restrict__ node_max, int* __restrict__ node_idx) {
  const int lane = threadIdx.x & 31;
  const int wave = threadIdx.x >> 5;
  const int batch = blockIdx.x >> 5;            // 32 blocks per batch
  const int strip = ((blockIdx.x & 31) << 3) + wave;
  const int m0 = strip * 16;
  const int h = lane >> 4;
  const int lrow = lane & 15;

  const _Float16* arow = am + ((size_t)batch * HALF_T + (m0 + lrow)) * CH;
  const v16h A0 = load_opnd(arow, 0, h);
  const v16h A1 = load_opnd(arow, 1, h);
  const v16h A2 = load_opnd(arow, 2, h);
  const v16h A3 = load_opnd(arow, 3, h);

  float rmax[8];
  int   ridx[8];
  #pragma unroll
  for (int g = 0; g < 8; ++g) { rmax[g] = -3.402823466e38f; ridx[g] = 0; }

  const _Float16* bbase = bm + (size_t)batch * HALF_T * CH;

  // double-buffered B tile registers
  v16h B0[2], B1[2], B2[2], B3[2];
  {
    const _Float16* brow = bbase + (size_t)lrow * CH;   // tile 0
    B0[0] = load_opnd(brow, 0, h);
    B1[0] = load_opnd(brow, 1, h);
    B2[0] = load_opnd(brow, 2, h);
    B3[0] = load_opnd(brow, 3, h);
  }

  #pragma unroll 2
  for (int nt = 0; nt < NTILES; ++nt) {
    const int cur = nt & 1;
    const int nxt = cur ^ 1;
    if (nt + 1 < NTILES) {                        // uniform branch: EXEC stays full
      const _Float16* brow = bbase + (size_t)((nt + 1) * 16 + lrow) * CH;
      if (nt + 2 < NTILES)                        // WGP-scope prefetch, 2 tiles ahead
        __builtin_prefetch(brow + 16 * CH, 0, 3);
      B0[nxt] = load_opnd(brow, 0, h);
      B1[nxt] = load_opnd(brow, 1, h);
      B2[nxt] = load_opnd(brow, 2, h);
      B3[nxt] = load_opnd(brow, 3, h);
    }

    v8f c = {};
    c = __builtin_amdgcn_wmma_f32_16x16x32_f16(false, A0, false, B0[cur],
                                               (short)0, c, false, false);
    c = __builtin_amdgcn_wmma_f32_16x16x32_f16(false, A1, false, B1[cur],
                                               (short)0, c, false, false);
    c = __builtin_amdgcn_wmma_f32_16x16x32_f16(false, A2, false, B2[cur],
                                               (short)0, c, false, false);
    c = __builtin_amdgcn_wmma_f32_16x16x32_f16(false, A3, false, B3[cur],
                                               (short)0, c, false, false);

    // C/D layout: VGPR g, lanes 0-15 -> (row m0+g,   col nt*16+lane)
    //                       lanes 16-31 -> (row m0+8+g, col nt*16+lane-16)
    const int ncol = nt * 16 + lrow;
    #pragma unroll
    for (int g = 0; g < 8; ++g) {
      const float s = c[g];
      if (s > rmax[g]) { rmax[g] = s; ridx[g] = ncol; }  // strict > keeps first
    }
  }

  // butterfly over each 16-lane half (cols); tie -> smaller index
  #pragma unroll
  for (int g = 0; g < 8; ++g) {
    #pragma unroll
    for (int off = 1; off <= 8; off <<= 1) {
      const float om = __shfl_xor(rmax[g], off, 32);
      const int   oi = __shfl_xor(ridx[g], off, 32);
      if (om > rmax[g] || (om == rmax[g] && oi < ridx[g])) {
        rmax[g] = om; ridx[g] = oi;
      }
    }
  }

  if (lrow == 0) {
    const int rowbase = m0 + h * 8;  // lane0 -> rows m0..m0+7, lane16 -> +8..+15
    #pragma unroll
    for (int g = 0; g < 8; ++g) {
      node_max[(size_t)batch * HALF_T + rowbase + g] = rmax[g];
      node_idx[(size_t)batch * HALF_T + rowbase + g] = ridx[g];
    }
  }
}

// ---------------- 3) per-batch bitonic argsort (desc key, asc idx) ----------
__global__ void __launch_bounds__(1024)
sort_kernel(const float* __restrict__ node_max, int* __restrict__ edge_idx) {
  __shared__ float key[HALF_T];
  __shared__ int   idx[HALF_T];
  const int b = blockIdx.x;
  for (int i = threadIdx.x; i < HALF_T; i += 1024) {
    key[i] = node_max[(size_t)b * HALF_T + i];
    idx[i] = i;
  }
  __syncthreads();
  for (int k = 2; k <= HALF_T; k <<= 1) {
    for (int j = k >> 1; j > 0; j >>= 1) {
      for (int i = threadIdx.x; i < HALF_T; i += 1024) {
        const int ixj = i ^ j;
        if (ixj > i) {
          const float ka = key[i], kb = key[ixj];
          const int   ia = idx[i], ib = idx[ixj];
          // "b comes first": larger key, or equal key with smaller index
          const bool firstB = (kb > ka) || (kb == ka && ib < ia);
          const bool up = ((i & k) == 0);
          if (up ? firstB : !firstB) {
            key[i] = kb; key[ixj] = ka;
            idx[i] = ib; idx[ixj] = ia;
          }
        }
      }
      __syncthreads();
    }
  }
  for (int i = threadIdx.x; i < HALF_T; i += 1024)
    edge_idx[(size_t)b * HALF_T + i] = idx[i];
}

// ---------------- 4) out[:, :2048] = unm gather; out[:, 2048:] = dst_tok ----
__global__ void __launch_bounds__(128)
gather_init_kernel(const float* __restrict__ pts,
                   const int* __restrict__ edge_idx,
                   float* __restrict__ out, float* __restrict__ counts) {
  const int row = blockIdx.x;       // 0 .. B*OUT_T-1
  const int b = row / OUT_T;
  const int t = row % OUT_T;
  const int c = threadIdx.x;
  const float* src;
  if (t < UNM_CNT) {
    const int ui = edge_idx[(size_t)b * HALF_T + R_MERGE + t];
    src = pts + ((size_t)b * T_TOK + 2 * ui) * CH;        // even tokens
  } else {
    const int n = t - UNM_CNT;
    src = pts + ((size_t)b * T_TOK + 2 * n + 1) * CH;     // odd tokens
    if (c == 0) counts[(size_t)b * HALF_T + n] = 1.0f;
  }
  out[(size_t)row * CH + c] = src[c];
}

// ---------------- 5) scatter-add src rows into dst rows ----------------
__global__ void __launch_bounds__(128)
scatter_kernel(const float* __restrict__ pts,
               const int* __restrict__ edge_idx,
               const int* __restrict__ node_idx,
               float* __restrict__ out, float* __restrict__ counts) {
  const int j = blockIdx.x;         // 0 .. B*R_MERGE-1
  const int b = j / R_MERGE;
  const int s = j % R_MERGE;
  const int c = threadIdx.x;
  const int si = edge_idx[(size_t)b * HALF_T + s];
  const int di = node_idx[(size_t)b * HALF_T + si];
  const float v = pts[((size_t)b * T_TOK + 2 * si) * CH + c];
  atomicAdd(&out[((size_t)b * OUT_T + UNM_CNT + di) * CH + c], v);
  if (c == 0) atomicAdd(&counts[(size_t)b * HALF_T + di], 1.0f);
}

// ---------------- 6) divide dst rows by counts ----------------
__global__ void __launch_bounds__(128)
divide_kernel(float* __restrict__ out, const float* __restrict__ counts) {
  const int row = blockIdx.x;       // 0 .. B*HALF_T-1
  const int b = row / HALF_T;
  const int n = row % HALF_T;
  const int c = threadIdx.x;
  const float cnt = counts[(size_t)b * HALF_T + n];
  out[((size_t)b * OUT_T + UNM_CNT + n) * CH + c] /= cnt;
}

// ---------------- launch ----------------
extern "C" void kernel_launch(void* const* d_in, const int* in_sizes, int n_in,
                              void* d_out, int out_size, void* d_ws,
                              size_t ws_size, hipStream_t stream) {
  const float* pts = (const float*)d_in[0];
  float* out = (float*)d_out;
  char* ws = (char*)d_ws;

  const size_t panel_bytes = (size_t)BATCHES * HALF_T * CH * sizeof(_Float16); // 8 MB
  const size_t vec_bytes   = (size_t)BATCHES * HALF_T * sizeof(float);         // 128 KB

  _Float16* am       = (_Float16*)(ws);
  _Float16* bm       = (_Float16*)(ws + panel_bytes);
  float*    node_max = (float*)   (ws + 2 * panel_bytes);
  int*      node_idx = (int*)     (ws + 2 * panel_bytes + 1 * vec_bytes);
  int*      edge_idx = (int*)     (ws + 2 * panel_bytes + 2 * vec_bytes);
  float*    counts   = (float*)   (ws + 2 * panel_bytes + 3 * vec_bytes);

  normalize_kernel<<<(BATCHES * T_TOK) / 8, 256, 0, stream>>>(pts, am, bm);
  score_argmax_kernel<<<BATCHES * 32, 256, 0, stream>>>(am, bm, node_max,
                                                        node_idx);
  sort_kernel<<<BATCHES, 1024, 0, stream>>>(node_max, edge_idx);
  gather_init_kernel<<<BATCHES * OUT_T, 128, 0, stream>>>(pts, edge_idx, out,
                                                          counts);
  scatter_kernel<<<BATCHES * R_MERGE, 128, 0, stream>>>(pts, edge_idx,
                                                        node_idx, out, counts);
  divide_kernel<<<BATCHES * HALF_T, 128, 0, stream>>>(out, counts);
}
